// RNN_s_1176821039406
// MI455X (gfx1250) — matrix operand order repaired and checked by reading, not compiled, not run on hardware
//
#include <hip/hip_runtime.h>
#include <hip/hip_bf16.h>
#include <math.h>

typedef __attribute__((ext_vector_type(16))) _Float16 v16h;
typedef __attribute__((ext_vector_type(8)))  float    v8f;

#define TSTEPS 1301
#define SUBSEQ 100
#define HDIM   256
#define IDIM   64
#define ODIM   10
#define BATCH  128
#define BT     16    // batch rows per workgroup
#define NWGS   (BATCH / BT)   // 8 workgroups

// Build an f16 WMMA A-operand (16x32 MxK tile) from an LDS row-major f16 buffer.
// Layout (ISA 7.12.2): lanes 0-15 hold row M=lane, K = kb*32 + {0..7, 16..23};
// lanes 16-31 hold row M=lane-16, K = kb*32 + {8..15, 24..31}.
__device__ __forceinline__ v16h ldsA16(const _Float16* base, int m, int stride,
                                       int kb, int lane) {
  const _Float16* p = base + m * stride + kb * 32 + ((lane < 16) ? 0 : 8);
  v16h a;
#pragma unroll
  for (int i = 0; i < 8; i++) { a[i] = p[i]; a[8 + i] = p[16 + i]; }
  return a;
}

// Build an f16 WMMA B-operand (32x16 KxN tile) from a row-major f32 weight
// matrix W[N_total][K_total], where B[k][n] = W[n][k] (i.e. x @ W^T).
// Layout: lanes 0-15 hold column N=lane, K = kb*32 + {0..7, 16..23};
// lanes 16-31 hold column N=lane-16, K = kb*32 + {8..15, 24..31}.
__device__ __forceinline__ v16h mkB(const float* w, int col, int ktot, int kb,
                                    int lane, bool valid) {
  const float* p = w + (size_t)col * ktot + kb * 32 + ((lane < 16) ? 0 : 8);
  v16h b;
#pragma unroll
  for (int i = 0; i < 8; i++) {
    b[i] = valid ? (_Float16)p[i] : (_Float16)0.0f;
    b[8 + i] = valid ? (_Float16)p[16 + i] : (_Float16)0.0f;
  }
  return b;
}

__global__ __launch_bounds__(512) void snn_scan_kernel(
    const float* __restrict__ x,        // [128,1301,64]
    const int*   __restrict__ labels,   // [128,1301]
    const float* __restrict__ w_i2h,    // [256,64]
    const float* __restrict__ b_i2h,    // [256]
    const float* __restrict__ w_h2h,    // [256,256]
    const float* __restrict__ b_h2h,    // [256]
    const float* __restrict__ w_h2o,    // [10,256]
    const float* __restrict__ b_h2o,    // [10]
    const float* __restrict__ tau_m_h,  // [256]
    const float* __restrict__ tau_m_o,  // [10]
    const float* __restrict__ tau_adp_o,// [10]
    const float* __restrict__ h0,       // [128,256]
    const float* __restrict__ o0,       // [128,10]
    float* __restrict__ preds,          // [1201,128,10]
    float* __restrict__ partials)       // [16] : [0..8) loss, [8..16) spikes
{
  __shared__ _Float16 xLds[BT * IDIM];     // current x tile (f16)
  __shared__ _Float16 spkLds[BT * HDIM];   // hidden spikes (f16)
  __shared__ float oInL[BT * 16];
  __shared__ float oMemL[BT * 16];
  __shared__ float oSpkL[BT * 16];
  __shared__ float bOL[BT * 16];
  __shared__ float red[512];

  const int tid    = threadIdx.x;
  const int wave   = tid >> 5;
  const int lane   = tid & 31;
  const int ln     = lane & 15;
  const int hiHalf = (lane >= 16) ? 8 : 0;
  const int wg     = blockIdx.x;

  const int colN = wave * 16 + ln;         // hidden column owned by this lane

  // ---- Persistent f16 weight tiles in VGPRs (B operands) ----
  v16h Bi[2], Bh[8], Bo[8];
#pragma unroll
  for (int kb = 0; kb < 2; kb++) Bi[kb] = mkB(w_i2h, colN, IDIM, kb, lane, true);
#pragma unroll
  for (int kb = 0; kb < 8; kb++) Bh[kb] = mkB(w_h2h, colN, HDIM, kb, lane, true);
  if (wave == 0) {
#pragma unroll
    for (int kb = 0; kb < 8; kb++)
      Bo[kb] = mkB(w_h2o, (ln < ODIM) ? ln : 0, HDIM, kb, lane, ln < ODIM);
  }

  const float alphaH = expf(-1.0f / tau_m_h[colN]);
  const float biasH  = b_i2h[colN] + b_h2h[colN];
  const float biasO  = (ln < ODIM) ? b_h2o[ln] : 0.0f;

  // ---- Hidden state in C/D tile layout (lane (r): M = r+hiHalf, N = colN) ----
  float hMem[8], hSpk[8];
#pragma unroll
  for (int r = 0; r < 8; r++) {
    int M = r + hiHalf;
    float v = h0[(size_t)(wg * BT + M) * HDIM + colN];
    hMem[r] = v; hSpk[r] = v;
    spkLds[M * HDIM + colN] = (_Float16)v;
  }

  // ---- Output-layer state (160 elements) in LDS ----
  float alphaO = 0.0f, roO = 0.0f;
  if (tid < BT * ODIM) {
    int b = tid / ODIM, j = tid % ODIM;
    alphaO = expf(-1.0f / tau_m_o[j]);
    roO    = expf(-1.0f / tau_adp_o[j]);
    float v = o0[(size_t)(wg * BT + b) * ODIM + j];
    oMemL[b * 16 + j] = v; oSpkL[b * 16 + j] = v; bOL[b * 16 + j] = 0.01f;
  }

  float spkAcc = 0.0f;   // local contribution to total_spikes
  float nllAcc = 0.0f;   // local contribution to loss (threads < 16 only)

  for (int t = 0; t < TSTEPS; t++) {
    // Stage x_t tile into LDS as f16 (1024 elements, 2 per thread).
#pragma unroll
    for (int q = 0; q < 2; q++) {
      int e = tid * 2 + q;
      int m = e >> 6, k = e & 63;
      xLds[e] = (_Float16)x[((size_t)(wg * BT + m) * TSTEPS + t) * IDIM + k];
      if (t + 1 < TSTEPS)
        __builtin_prefetch(&x[((size_t)(wg * BT + m) * TSTEPS + t + 1) * IDIM + k], 0, 1);
    }
    __syncthreads();   // x_t staged; previous spikes visible

    // h_in = bias + x_t @ w_i2h^T + h_spk @ w_h2h^T   (per-wave 16x16 tile)
    v8f c;
#pragma unroll
    for (int r = 0; r < 8; r++) c[r] = biasH;
#pragma unroll
    for (int kb = 0; kb < 2; kb++) {
      v16h a = ldsA16(xLds, ln, IDIM, kb, lane);
      c = __builtin_amdgcn_wmma_f32_16x16x32_f16(false, a, false, Bi[kb],
                                                 (short)0, c, false, false);
    }
#pragma unroll
    for (int kb = 0; kb < 8; kb++) {
      v16h a = ldsA16(spkLds, ln, HDIM, kb, lane);
      c = __builtin_amdgcn_wmma_f32_16x16x32_f16(false, a, false, Bh[kb],
                                                 (short)0, c, false, false);
    }
    __syncthreads();   // all spkLds reads complete before overwrite

    // Hidden LIF update (beta = 0 -> B == b_j0 == 0.01)
#pragma unroll
    for (int r = 0; r < 8; r++) {
      float mem = hMem[r] * alphaH + (1.0f - alphaH) * c[r] - 0.01f * hSpk[r];
      float s = (mem - 0.01f) > 0.0f ? 1.0f : 0.0f;
      hMem[r] = mem; hSpk[r] = s; spkAcc += s;
      spkLds[(r + hiHalf) * HDIM + colN] = (_Float16)s;
    }
    __syncthreads();   // new spikes visible

    // o_in = h_spk_new @ w_h2o^T + b_h2o   (wave 0 only; full EXEC in-wave)
    if (wave == 0) {
      v8f co;
#pragma unroll
      for (int r = 0; r < 8; r++) co[r] = biasO;
#pragma unroll
      for (int kb = 0; kb < 8; kb++) {
        v16h a = ldsA16(spkLds, ln, HDIM, kb, lane);
        co = __builtin_amdgcn_wmma_f32_16x16x32_f16(false, a, false, Bo[kb],
                                                    (short)0, co, false, false);
      }
#pragma unroll
      for (int r = 0; r < 8; r++) oInL[(r + hiHalf) * 16 + ln] = co[r];
    }
    __syncthreads();   // o_in ready

    // Adaptive output update (beta = 1.8), 160 scalar elements
    if (tid < BT * ODIM) {
      int b = tid / ODIM, j = tid % ODIM;
      float bo  = roO * bOL[b * 16 + j] + (1.0f - roO) * oSpkL[b * 16 + j];
      float Bv  = 0.01f + 1.8f * bo;
      float mem = oMemL[b * 16 + j] * alphaO +
                  (1.0f - alphaO) * oInL[b * 16 + j] - Bv * oSpkL[b * 16 + j];
      float s = (mem - Bv) > 0.0f ? 1.0f : 0.0f;
      bOL[b * 16 + j] = bo; oMemL[b * 16 + j] = mem; oSpkL[b * 16 + j] = s;
      spkAcc += s;
    }
    __syncthreads();   // output state updated

    // log-softmax + NLL per batch row (10-wide, one thread per row)
    if (tid < BT) {
      int b = tid;
      float v[ODIM]; float mx = -3.4e38f;
#pragma unroll
      for (int j = 0; j < ODIM; j++) { v[j] = oMemL[b * 16 + j]; mx = v[j] > mx ? v[j] : mx; }
      float se = 0.0f;
#pragma unroll
      for (int j = 0; j < ODIM; j++) se += expf(v[j] - mx);
      float ls = mx + logf(se);
      if (t >= SUBSEQ) {
        size_t rowBase = ((size_t)(t - SUBSEQ) * BATCH + wg * BT + b) * ODIM;
#pragma unroll
        for (int j = 0; j < ODIM; j++) preds[rowBase + j] = v[j] - ls;
        int lab = labels[(size_t)(wg * BT + b) * TSTEPS + t];
        nllAcc += (ls - v[lab]) * (1.0f / 128.0f);
      }
    }
    __syncthreads();
  }

  // ---- Reductions: spikes over all 512 threads, nll over first 16 ----
  red[tid] = spkAcc;
  __syncthreads();
  for (int s = 256; s > 0; s >>= 1) {
    if (tid < s) red[tid] += red[tid + s];
    __syncthreads();
  }
  if (tid == 0) partials[8 + wg] = red[0];
  __syncthreads();

  red[tid] = (tid < BT) ? nllAcc : 0.0f;
  __syncthreads();
  for (int s = 256; s > 0; s >>= 1) {
    if (tid < s) red[tid] += red[tid + s];
    __syncthreads();
  }
  if (tid == 0) partials[wg] = red[0];
}

__global__ void snn_finalize_kernel(const float* __restrict__ partials,
                                    float* __restrict__ out, int out_size) {
  if (blockIdx.x == 0 && threadIdx.x == 0) {
    float loss = 0.0f, spk = 0.0f;
    for (int i = 0; i < NWGS; i++) { loss += partials[i]; spk += partials[8 + i]; }
    out[out_size - 2] = loss;   // loss follows the 1201*128*10 predictions
    out[out_size - 1] = spk;    // total_spikes is last
  }
}

extern "C" void kernel_launch(void* const* d_in, const int* in_sizes, int n_in,
                              void* d_out, int out_size, void* d_ws, size_t ws_size,
                              hipStream_t stream) {
  const float* x        = (const float*)d_in[0];
  const int*   labels   = (const int*)  d_in[1];
  const float* w_i2h    = (const float*)d_in[2];
  const float* b_i2h    = (const float*)d_in[3];
  const float* w_h2h    = (const float*)d_in[4];
  const float* b_h2h    = (const float*)d_in[5];
  const float* w_h2o    = (const float*)d_in[6];
  const float* b_h2o    = (const float*)d_in[7];
  const float* tau_m_h  = (const float*)d_in[8];
  // d_in[9] = tau_adp_h: unused (beta==0 for hidden layer => B == b_j0)
  const float* tau_m_o  = (const float*)d_in[10];
  const float* tau_adp_o= (const float*)d_in[11];
  const float* h0       = (const float*)d_in[12];
  const float* o0       = (const float*)d_in[13];

  float* out      = (float*)d_out;
  float* partials = (float*)d_ws;   // 16 floats of scratch

  snn_scan_kernel<<<NWGS, 512, 0, stream>>>(
      x, labels, w_i2h, b_i2h, w_h2h, b_h2h, w_h2o, b_h2o,
      tau_m_h, tau_m_o, tau_adp_o, h0, o0, out, partials);
  snn_finalize_kernel<<<1, 32, 0, stream>>>(partials, out, out_size);
}
